// ConceptGraph_62740882260557
// MI455X (gfx1250) — matrix operand order repaired and checked by reading, not compiled, not run on hardware
//
#include <hip/hip_runtime.h>
#include <hip/hip_bf16.h>

typedef __attribute__((ext_vector_type(16))) _Float16 v16h;
typedef __attribute__((ext_vector_type(8)))  _Float16 v8h;
typedef __attribute__((ext_vector_type(4)))  _Float16 v4h;
typedef __attribute__((ext_vector_type(8)))  float    v8f;

#define D_DIM 1024
#define MT 128          // x rows per block
#define NT 128          // codes per accumulation chunk
#define KC 64           // d elements staged per LDS pass
#define AS 72           // LDS row stride in halves (64 + 8 pad; 144B = conflict-free)

typedef union { v16h v; v8h h[2]; } Frag;

// ---------------- Kernel 1: codebook squared norms -----------------
__global__ void vq_enorm_kernel(const float* __restrict__ E, float* __restrict__ enorm) {
    const int code = blockIdx.x * 8 + (threadIdx.x >> 5);
    const int lane = threadIdx.x & 31;
    const float4* row = (const float4*)(E + (size_t)code * D_DIM);
    float s = 0.f;
    #pragma unroll 4
    for (int i = lane; i < D_DIM / 4; i += 32) {
        float4 v = row[i];
        s += v.x * v.x + v.y * v.y + v.z * v.z + v.w * v.w;
    }
    #pragma unroll
    for (int off = 16; off; off >>= 1) s += __shfl_xor(s, off, 32);
    if (lane == 0) enorm[code] = s;
}

// ---------------- Kernel 2: WMMA distance GEMM + streaming argmin ----
__global__ __launch_bounds__(256)
void vq_argmin_kernel(const float* __restrict__ X,
                      const float* __restrict__ E,
                      const float* __restrict__ enorm,
                      int* __restrict__ idx_out,
                      int nCodes) {
    __shared__ _Float16 ldsA[MT * AS];
    __shared__ _Float16 ldsB[NT * AS];

    const int tid    = threadIdx.x;
    const int lane   = tid & 31;
    const int wave   = tid >> 5;            // 0..7, owns rows wave*16..+15
    const int rowBlk = blockIdx.x * MT;
    const int col    = lane & 15;           // N / M position within a tile
    const int hiHalf = (lane >= 16);        // lanes 16..31

    // tile loader mapping: 16 threads per row, float4 each -> 64 floats/row
    const int lr = tid >> 4;                // 0..15
    const int lc = (tid & 15) * 4;          // float col 0,4,...,60

    // uniform SGPR base for x rows of this block; per-lane 32-bit offsets
    const float* __restrict__ Xb = X + (size_t)rowBlk * D_DIM;

    float minv[8];
    int   mini[8];
    #pragma unroll
    for (int r = 0; r < 8; ++r) { minv[r] = 3.4e38f; mini[r] = 0; }

    for (int chunk = 0; chunk < nCodes; chunk += NT) {
        const float* __restrict__ Eb = E + (size_t)chunk * D_DIM;   // uniform per chunk
        const float* __restrict__ nb = enorm + chunk;

        v8f acc[8];
        #pragma unroll
        for (int t = 0; t < 8; ++t) acc[t] = {};

        for (int dc = 0; dc < D_DIM; dc += KC) {
            __syncthreads();
            // stage 128x64 A (x rows) and B (codebook rows) tiles, fp32 -> f16
            #pragma unroll
            for (int p = 0; p < 8; ++p) {
                const int row = p * 16 + lr;
                const int off = row * D_DIM + dc + lc;              // 32-bit GVS offset
                const float4 a = *(const float4*)(Xb + off);
                v4h ha; ha.x = (_Float16)a.x; ha.y = (_Float16)a.y;
                        ha.z = (_Float16)a.z; ha.w = (_Float16)a.w;
                *(v4h*)&ldsA[row * AS + lc] = ha;
                const float4 b = *(const float4*)(Eb + off);
                v4h hb; hb.x = (_Float16)b.x; hb.y = (_Float16)b.y;
                        hb.z = (_Float16)b.z; hb.w = (_Float16)b.w;
                *(v4h*)&ldsB[row * AS + lc] = hb;
            }
            // hint next KC slice (rows 0..15 of this E chunk) toward the caches (global_prefetch_b8)
            if (dc + KC < D_DIM)
                __builtin_prefetch(Eb + lr * D_DIM + dc + KC + lc, 0, 0);
            __syncthreads();

            // hoist both k-step A fragments (16x32 f16 each, reused by 8 WMMAs)
            Frag a0, a1;
            {
                const _Float16* ar = &ldsA[(wave * 16 + col) * AS + (hiHalf ? 16 : 0)];
                a0.h[0] = *(const v8h*)(ar);      a0.h[1] = *(const v8h*)(ar + 8);
                a1.h[0] = *(const v8h*)(ar + 32); a1.h[1] = *(const v8h*)(ar + 40);
            }
            // depth-2 software pipeline over the 16 (ks,t) fragments:
            // at step i, issue loads for fragment i+2, consume fragment i.
            const int bh = hiHalf ? 16 : 0;
            const _Float16* __restrict__ Bf = &ldsB[col * AS + bh];
            Frag b[3];
            #pragma unroll
            for (int i = 0; i < 2; ++i) {
                const _Float16* p = Bf + (i * 16) * AS;     // ks=0, t=i
                b[i].h[0] = *(const v8h*)(p);
                b[i].h[1] = *(const v8h*)(p + 8);
            }
            #pragma unroll
            for (int i = 0; i < 16; ++i) {                  // i = ks*8 + t
                if (i + 2 < 16) {
                    const int j  = i + 2;
                    const _Float16* p = Bf + ((j & 7) * 16) * AS + (j >> 3) * 32;
                    b[j % 3].h[0] = *(const v8h*)(p);
                    b[j % 3].h[1] = *(const v8h*)(p + 8);
                }
                acc[i & 7] = __builtin_amdgcn_wmma_f32_16x16x32_f16(
                    false, (i < 8) ? a0.v : a1.v,
                    false, b[i % 3].v,
                    (short)0, acc[i & 7], false, false);
            }
        }

        // epilogue: score = ||e||^2 - 2*dot ; streaming per-lane argmin
        #pragma unroll
        for (int t = 0; t < 8; ++t) {
            const int c16 = t * 16 + col;
            const float nrm = nb[c16];
            const int code = chunk + c16;
            #pragma unroll
            for (int r = 0; r < 8; ++r) {
                const float score = fmaf(-2.0f, acc[t][r], nrm);
                if (score < minv[r]) { minv[r] = score; mini[r] = code; }
            }
        }
    }

    // cross-lane reduce over the 16 columns (C/D layout: vgpr r = row r / r+8)
    #pragma unroll
    for (int r = 0; r < 8; ++r) {
        float v = minv[r];
        int   i = mini[r];
        #pragma unroll
        for (int off = 8; off; off >>= 1) {
            const float ov = __shfl_xor(v, off, 16);
            const int   oi = __shfl_xor(i, off, 16);
            if (ov < v || (ov == v && oi < i)) { v = ov; i = oi; }
        }
        if (col == 0) {
            const int mrow = wave * 16 + (hiHalf ? 8 : 0) + r;
            idx_out[rowBlk + mrow] = i;
        }
    }
}

// ---------------- Kernel 3: gather winning rows (fp32 stream) --------
__global__ void vq_gather_kernel(const float* __restrict__ E,
                                 const int* __restrict__ idx,
                                 float* __restrict__ out) {
    const int row = blockIdx.x;
    const int k   = idx[row];
    const float4* src = (const float4*)(E + (size_t)k * D_DIM);
    float4* dst = (float4*)(out + (size_t)row * D_DIM);
    dst[threadIdx.x] = src[threadIdx.x];   // 256 threads * 16B = 4KB row
}

extern "C" void kernel_launch(void* const* d_in, const int* in_sizes, int n_in,
                              void* d_out, int out_size, void* d_ws, size_t ws_size,
                              hipStream_t stream) {
    const float* x = (const float*)d_in[0];   // [B*T, 1024] fp32
    const float* E = (const float*)d_in[1];   // [K, 1024] fp32
    float* out = (float*)d_out;               // [B*T, 1024] fp32

    const int nRows  = in_sizes[0] / D_DIM;   // 16384
    const int nCodes = in_sizes[1] / D_DIM;   // 4096

    float* enorm = (float*)d_ws;
    int*   idx   = (int*)((char*)d_ws + (((size_t)nCodes * sizeof(float) + 255) & ~(size_t)255));

    vq_enorm_kernel<<<nCodes / 8, 256, 0, stream>>>(E, enorm);
    vq_argmin_kernel<<<nRows / MT, 256, 0, stream>>>(x, E, enorm, idx, nCodes);
    vq_gather_kernel<<<nRows, 256, 0, stream>>>(E, idx, out);
}
